// GAT_62405874811847
// MI455X (gfx1250) — compile-verified, matrix operand order checked by reading, not run on hardware
//
#include <hip/hip_runtime.h>

#define BB 2
#define NN 2048
#define FF 256
#define HH 8
#define OO 32
#define BN (BB*NN)
#define ALPHA 0.2f
#define NEGV (-1.0e15f)

typedef __attribute__((ext_vector_type(16))) _Float16 v16h;
typedef __attribute__((ext_vector_type(8)))  _Float16 v8h;
typedef __attribute__((ext_vector_type(8)))  float    v8f;
typedef __attribute__((ext_vector_type(4)))  float    v4f;
typedef __attribute__((ext_vector_type(4)))  int      v4i;

static __device__ __forceinline__ v16h cat8(v8h lo, v8h hi) {
  v16h r;
#pragma unroll
  for (int i = 0; i < 8; ++i) { r[i] = lo[i]; r[8 + i] = hi[i]; }
  return r;
}

static __device__ __forceinline__ float eluf(float x) {
  return x > 0.f ? x : (__expf(x) - 1.f);
}

// ---------------- elementwise f32 -> f16 convert ----------------
__global__ void convert_f16(const float* __restrict__ src, _Float16* __restrict__ dst, int n) {
  int i = blockIdx.x * blockDim.x + threadIdx.x;
  if (i < n) dst[i] = (_Float16)src[i];
}

// ---------------- build transposed f16 weight [col=256][k=256] ----------------
// mode 0: W_heads[l] laid out [H,F,O] -> col = h*O+o ; mode 1: W_out[l] [F,F] -> col = c
__global__ void build_w16t(const float* __restrict__ w, _Float16* __restrict__ wt, int mode) {
  int idx = blockIdx.x * blockDim.x + threadIdx.x;   // 65536 = 256 cols * 256 k
  int c = idx >> 8, f = idx & 255;
  float v;
  if (mode == 0) { int hh = c >> 5, o = c & 31; v = w[hh * (FF * OO) + f * OO + o]; }
  else           { v = w[f * FF + c]; }
  wt[idx] = (_Float16)v;
}

// ---------------- projection GEMM: C[4096x256] = A[4096x256] * W[256x256] ----------------
// Bt is W transposed [c][k]. Writes C (f32, row-major) and Ct (f16, transposed [c][row]).
__global__ void gemm16(const _Float16* __restrict__ A, const _Float16* __restrict__ Bt,
                       float* __restrict__ C, _Float16* __restrict__ Ct) {
  int bx = blockIdx.x;
  int ct = bx & 7, rt = bx >> 3;
  int rbase = rt << 4, cb = ct << 5;
  int lane = threadIdx.x;
  int m = lane & 15, g = lane >> 4;
  const _Float16* ar  = A  + (size_t)(rbase + m) * FF;
  const _Float16* b0r = Bt + (size_t)(cb + m) * FF;
  const _Float16* b1r = Bt + (size_t)(cb + 16 + m) * FF;
  v8f c0 = {}; v8f c1 = {};
#pragma unroll
  for (int kc = 0; kc < FF; kc += 32) {
    v16h a  = cat8(*(const v8h*)(ar  + kc + 8 * g), *(const v8h*)(ar  + kc + 16 + 8 * g));
    v16h b0 = cat8(*(const v8h*)(b0r + kc + 8 * g), *(const v8h*)(b0r + kc + 16 + 8 * g));
    v16h b1 = cat8(*(const v8h*)(b1r + kc + 8 * g), *(const v8h*)(b1r + kc + 16 + 8 * g));
    c0 = __builtin_amdgcn_wmma_f32_16x16x32_f16(false, a, false, b0, (short)0, c0, false, false);
    c1 = __builtin_amdgcn_wmma_f32_16x16x32_f16(false, a, false, b1, (short)0, c1, false, false);
  }
#pragma unroll
  for (int v = 0; v < 8; ++v) {
    int row = rbase + v + 8 * g;
    C [(size_t)row * FF + cb + m]        = c0[v];
    C [(size_t)row * FF + cb + 16 + m]   = c1[v];
    Ct[(size_t)(cb + m)      * BN + row] = (_Float16)c0[v];
    Ct[(size_t)(cb + 16 + m) * BN + row] = (_Float16)c1[v];
  }
}

// ---------------- attention scores s1,s2 = Wh . a1 / Wh . a2 ----------------
__global__ void score_kernel(const float* __restrict__ Wh, const float* __restrict__ a1b,
                             int a_stride, float* __restrict__ s1, float* __restrict__ s2,
                             int Hp, int Op) {
  int idx = blockIdx.x * blockDim.x + threadIdx.x;
  if (idx >= BB * Hp * NN) return;
  int n = idx % NN; int t = idx / NN; int hh = t % Hp; int b = t / Hp;
  const float* wr = Wh + ((size_t)(b * NN + n)) * FF + hh * Op;
  const float* a1 = a1b + hh * a_stride;
  const float* a2 = a1 + Op;
  float d1 = 0.f, d2 = 0.f;
  for (int o = 0; o < Op; ++o) { float w = wr[o]; d1 += w * a1[o]; d2 += w * a2[o]; }
  s1[idx] = d1; s2[idx] = d2;
}

// ---------------- per-row softmax statistics: rowM = max_j, rowInv = 1/sum_j ----------------
// One wave per row; lanes split j with contiguous v4 chunks (coalesced), butterfly merge.
__global__ void stats_kernel(const float* __restrict__ s1, const float* __restrict__ s2,
                             const int* __restrict__ adj, float* __restrict__ rowM,
                             float* __restrict__ rowInv, int Hp) {
  int gid = blockIdx.x * blockDim.x + threadIdx.x;
  int row = gid >> 5;
  int lane = gid & 31;
  if (row >= BB * Hp * NN) return;
  int n = row % NN; int t = row / NN; int hh = t % Hp; int b = t / Hp;
  float s1i = s1[row];
  const float* sp = s2 + (size_t)(b * Hp + hh) * NN;
  const int*   ap = adj + (size_t)b * NN * NN + (size_t)n * NN;
  float mx = -3.0e38f, sm = 0.f;
  for (int k = 0; k < NN; k += 128) {
    int j = k + lane * 4;
    v4f sv = *(const v4f*)(sp + j);
    v4i av = *(const v4i*)(ap + j);
#pragma unroll
    for (int q = 0; q < 4; ++q) {
      float e = s1i + sv[q];
      e = e > 0.f ? e : ALPHA * e;
      e = av[q] > 0 ? e : NEGV;
      float nm = fmaxf(mx, e);
      sm = sm * __expf(mx - nm) + __expf(e - nm);
      mx = nm;
    }
  }
#pragma unroll
  for (int off = 16; off >= 1; off >>= 1) {
    float mo = __shfl_xor(mx, off, 32);
    float so = __shfl_xor(sm, off, 32);
    float nm = fmaxf(mx, mo);
    sm = sm * __expf(mx - nm) + so * __expf(mo - nm);
    mx = nm;
  }
  if (lane == 0) { rowM[row] = mx; rowInv[row] = 1.0f / sm; }
}

// ---------------- fused attention P @ V with WMMA (+ELU), stats precomputed ----------------
// Block = one wave = (b, head, 16-row tile, NT*16-col o-slice).
template<int NT>
__global__ void gat_attn2(const float* __restrict__ s1, const float* __restrict__ s2,
                          const int* __restrict__ adj, const _Float16* __restrict__ Vt,
                          const float* __restrict__ rowM, const float* __restrict__ rowInv,
                          float* __restrict__ dst, int Hp, int Op, int oslices, int elu2) {
  int bx = blockIdx.x;
  int ot = bx % oslices; int t = bx / oslices;
  int rt = t & 127; t >>= 7;
  int hh = t % Hp; int b = t / Hp;
  int i0 = rt << 4;
  int colbase = hh * Op + ot * (NT * 16);
  int lane = threadIdx.x;
  int m = lane & 15, g = lane >> 4;

  int rowbase = (b * Hp + hh) * NN;
  const float* s2b = s2 + (size_t)rowbase;
  const int* arow = adj + (size_t)b * NN * NN + (size_t)(i0 + m) * NN;
  float s1i = s1[rowbase + i0 + m];
  float M   = rowM[rowbase + i0 + m];
  float inv = rowInv[rowbase + i0 + m];

  const _Float16* vp[NT];
#pragma unroll
  for (int tt = 0; tt < NT; ++tt)
    vp[tt] = Vt + (size_t)(colbase + tt * 16 + m) * BN + b * NN;
  v8f zero = {};
  v8f acc[NT];
#pragma unroll
  for (int tt = 0; tt < NT; ++tt) acc[tt] = zero;

  for (int jt = 0; jt < NN; jt += 32) {
    int jb0 = jt + 8 * g;
    int jb1 = jt + 16 + 8 * g;
    __builtin_prefetch(arow + jt + 256, 0, 1);
    __builtin_prefetch(s2b + jt + 256, 0, 1);
    v4f s2q0 = *(const v4f*)(s2b + jb0);
    v4f s2q1 = *(const v4f*)(s2b + jb0 + 4);
    v4f s2q2 = *(const v4f*)(s2b + jb1);
    v4f s2q3 = *(const v4f*)(s2b + jb1 + 4);
    v4i aq0 = *(const v4i*)(arow + jb0);
    v4i aq1 = *(const v4i*)(arow + jb0 + 4);
    v4i aq2 = *(const v4i*)(arow + jb1);
    v4i aq3 = *(const v4i*)(arow + jb1 + 4);
    v16h A;
#pragma unroll
    for (int q = 0; q < 4; ++q) {
      float e = s1i + s2q0[q]; e = e > 0.f ? e : ALPHA * e; e = aq0[q] > 0 ? e : NEGV;
      A[q]      = (_Float16)(__expf(e - M) * inv);
      e = s1i + s2q1[q]; e = e > 0.f ? e : ALPHA * e; e = aq1[q] > 0 ? e : NEGV;
      A[4 + q]  = (_Float16)(__expf(e - M) * inv);
      e = s1i + s2q2[q]; e = e > 0.f ? e : ALPHA * e; e = aq2[q] > 0 ? e : NEGV;
      A[8 + q]  = (_Float16)(__expf(e - M) * inv);
      e = s1i + s2q3[q]; e = e > 0.f ? e : ALPHA * e; e = aq3[q] > 0 ? e : NEGV;
      A[12 + q] = (_Float16)(__expf(e - M) * inv);
    }
#pragma unroll
    for (int tt = 0; tt < NT; ++tt) {
      v16h Bm = cat8(*(const v8h*)(vp[tt] + jb0), *(const v8h*)(vp[tt] + jb1));
      acc[tt] = __builtin_amdgcn_wmma_f32_16x16x32_f16(false, A, false, Bm, (short)0,
                                                       acc[tt], false, false);
    }
  }
#pragma unroll
  for (int tt = 0; tt < NT; ++tt) {
#pragma unroll
    for (int v = 0; v < 8; ++v) {
      int row = i0 + v + 8 * g;
      float x = eluf(acc[tt][v]);
      if (elu2) x = eluf(x);
      dst[(size_t)(b * NN + row) * FF + colbase + tt * 16 + m] = x;
    }
  }
}

extern "C" void kernel_launch(void* const* d_in, const int* in_sizes, int n_in,
                              void* d_out, int out_size, void* d_ws, size_t ws_size,
                              hipStream_t stream) {
  const float* x      = (const float*)d_in[0];
  const int*   adj    = (const int*)d_in[1];
  const float* Wheads = (const float*)d_in[2];
  const float* aheads = (const float*)d_in[3];
  const float* Wout   = (const float*)d_in[4];
  const float* aout   = (const float*)d_in[5];
  char* ws = (char*)d_ws;
  // workspace layout (~12.6 MB total)
  float*    h    = (float*)ws;                                       // 4 MB
  _Float16* h16  = (_Float16*)(ws + (4u  << 20));                    // 2 MB
  float*    Wh   = (float*)(ws + (6u  << 20));                       // 4 MB
  _Float16* WhT  = (_Float16*)(ws + (10u << 20));                    // 2 MB (transposed)
  float*    s1   = (float*)(ws + (12u << 20));                       // 128 KB
  float*    s2   = (float*)(ws + (12u << 20) + (128u << 10));        // 128 KB
  float*    rowM = (float*)(ws + (12u << 20) + (256u << 10));        // 128 KB
  float*    rowI = (float*)(ws + (12u << 20) + (384u << 10));        // 128 KB
  _Float16* W16T = (_Float16*)(ws + (12u << 20) + (512u << 10));     // 128 KB
  float* out = (float*)d_out;

  hipMemcpyAsync(h, x, sizeof(float) * BN * FF, hipMemcpyDeviceToDevice, stream);

  for (int l = 0; l < 3; ++l) {
    // ---- multi-head GAT (H=8, O=32) ----
    convert_f16 <<<4096, 256, 0, stream>>>(h, h16, BN * FF);
    build_w16t  <<<256, 256, 0, stream>>>(Wheads + (size_t)l * HH * FF * OO, W16T, 0);
    gemm16      <<<2048, 32, 0, stream>>>(h16, W16T, Wh, WhT);
    score_kernel<<<128, 256, 0, stream>>>(Wh, aheads + (size_t)l * HH * 2 * OO, 2 * OO,
                                          s1, s2, HH, OO);
    stats_kernel<<<4096, 256, 0, stream>>>(s1, s2, adj, rowM, rowI, HH);
    gat_attn2<2><<<2048, 32, 0, stream>>>(s1, s2, adj, WhT, rowM, rowI, h, HH, OO, 1, 0);
    // ---- output GAT (H=1, O=256), extra ELU ----
    convert_f16 <<<4096, 256, 0, stream>>>(h, h16, BN * FF);
    build_w16t  <<<256, 256, 0, stream>>>(Wout + (size_t)l * FF * FF, W16T, 1);
    gemm16      <<<2048, 32, 0, stream>>>(h16, W16T, Wh, WhT);
    score_kernel<<<16, 256, 0, stream>>>(Wh, aout + (size_t)l * 2 * FF, 0, s1, s2, 1, FF);
    stats_kernel<<<512, 256, 0, stream>>>(s1, s2, adj, rowM, rowI, 1);
    gat_attn2<4><<<1024, 32, 0, stream>>>(s1, s2, adj, WhT, rowM, rowI,
                                          (l == 2 ? out : h), 1, FF, 4, 1);
  }
}
